// PhysicsExpert_1382979469673
// MI455X (gfx1250) — matrix operand order repaired and checked by reading, not compiled, not run on hardware
//
#include <hip/hip_runtime.h>

// ---------------------------------------------------------------------------
// GNN message-passing step for MI455X (gfx1250, wave32, WMMA).
// bf16 activations/weights, fp32 accumulate via v_wmma_f32_16x16x32_bf16.
// ~84 GFLOP compute (94% in the edge MLP), ~1 GB HBM traffic.
// Edge kernel uses 32-row tiles (2 M-tiles per B fragment) to halve
// B-operand fetch per FLOP.
// ---------------------------------------------------------------------------

#define HID 128
#define WPB 4     // waves per block, node kernels (128 threads)
#define WPB_E 2   // waves per block, edge kernel (64 threads, big LDS tiles)

typedef __attribute__((ext_vector_type(16))) __bf16 v16bf;
typedef __attribute__((ext_vector_type(8)))  float  v8f;
typedef __attribute__((ext_vector_type(4)))  unsigned int v4u;

union FragU { v4u u4[2]; v16bf bf; };

__device__ __forceinline__ unsigned short f2bf(float x) {
  unsigned u = __float_as_uint(x);
  unsigned r = u + 0x7FFFu + ((u >> 16) & 1u);   // round-to-nearest-even
  return (unsigned short)(r >> 16);
}

// A fragment (16x32 bf16, M x K) from a row-major LDS tile.
// ISA layout: lane l holds row M=l%16; lanes 0-15 cover K=[kt*32+0..7] and
// [kt*32+16..23]; lanes 16-31 cover K=[+8..15] and [+24..31].
__device__ __forceinline__ v16bf afrag_lds(const unsigned short* tile,
                                           int strideElems, int kt, int lane) {
  int row  = lane & 15;
  int koff = kt * 32 + ((lane >> 4) << 3);
  FragU f;
  f.u4[0] = *(const v4u*)(tile + row * strideElems + koff);
  f.u4[1] = *(const v4u*)(tile + row * strideElems + koff + 16);
  return f.bf;
}

// B fragment (32x16 bf16, K x N) from pre-swizzled global weights:
// per fragment 32 lanes x 32B; lane loads two b128s.
__device__ __forceinline__ v16bf bfrag(const unsigned short* packed,
                                       int ntiles, int kt, int nt, int lane) {
  const unsigned short* p = packed + (((size_t)(kt * ntiles + nt)) << 9) + lane * 16;
  FragU f;
  f.u4[0] = *(const v4u*)(p);
  f.u4[1] = *(const v4u*)(p + 8);
  return f.bf;
}

__device__ __forceinline__ v8f wmma_bf16(v16bf a, v16bf b, v8f c) {
  return __builtin_amdgcn_wmma_f32_16x16x32_bf16(
      /*neg_a=*/false, a, /*neg_b=*/false, b,
      /*c_mod=*/(short)0, c, /*reuse_a=*/false, /*reuse_b=*/false);
}

// LDS ops are in-order per wave; this is a HW drain + compiler reorder fence
// for intra-wave LDS producer -> cross-lane consumer.
__device__ __forceinline__ void lds_fence() {
  asm volatile("s_wait_dscnt 0x0" ::: "memory");
}

// ---------------------------------------------------------------------------
// Pack fp32 weights W[K][N] (row stride rowStride) into bf16 B-fragment
// layout, padding K->Kpad, N->Npad with zeros. Element (k,n):
//   lane = (n%16) + 16*(((k%32)/8)%2), dword i = ((k%8) + 8*((k%32)/16))/2, half = k%2.
// ---------------------------------------------------------------------------
__global__ void pack_weights_kernel(const float* __restrict__ W, int K, int N,
                                    int rowStride, int Kpad, int Npad,
                                    unsigned short* __restrict__ dst) {
  int i = blockIdx.x * blockDim.x + threadIdx.x;
  int total = Kpad * Npad;
  if (i >= total) return;
  int k = i / Npad, n = i % Npad;
  int kt = k >> 5, nt = n >> 4, kk = k & 31, nn = n & 15;
  int khalf = (kk >> 3) & 1, ii = ((kk & 7) + ((kk >> 4) << 3)) >> 1, pair = kk & 1;
  int lane = nn + khalf * 16;
  int ntiles = Npad >> 4;
  size_t off = (((size_t)(kt * ntiles + nt)) << 9) + lane * 16 + ii * 2 + pair;
  float v = (k < K && n < N) ? W[(size_t)k * rowStride + n] : 0.0f;
  dst[off] = f2bf(v);
}

// ---------------------------------------------------------------------------
// node_encoder: h_v = relu(x @ W1 + b1) @ W2 + b2, stored as bf16 [N,128].
// One wave per 16-node tile.
// ---------------------------------------------------------------------------
__global__ void __launch_bounds__(128) node_encoder_kernel(
    const float* __restrict__ ns, const float* __restrict__ b1,
    const float* __restrict__ b2, const unsigned short* __restrict__ W1p,
    const unsigned short* __restrict__ W2p, unsigned short* __restrict__ hv,
    int nNodes) {
  __shared__ unsigned short inTile[WPB][16 * 32];
  __shared__ unsigned short hidTile[WPB][16 * 128];
  int wave = threadIdx.x >> 5, lane = threadIdx.x & 31;
  int tile = blockIdx.x * WPB + wave;
  if (tile * 16 >= nNodes) return;           // wave-uniform: EXEC stays full
  int base = tile * 16;

  if (lane < 16) {
    int r = base + lane;
    unsigned short* dstRow = inTile[wave] + lane * 32;
    if (r < nNodes) {
      const float* src = ns + (size_t)r * 9;
      #pragma unroll
      for (int j = 0; j < 9; ++j) dstRow[j] = f2bf(src[j]);
    } else {
      #pragma unroll
      for (int j = 0; j < 9; ++j) dstRow[j] = 0;
    }
    #pragma unroll
    for (int j = 9; j < 32; ++j) dstRow[j] = 0;
  }
  lds_fence();

  // layer 1: K=32 (1 ktile, 9 real), N=128 (8 ntiles)
  v16bf a0 = afrag_lds(inTile[wave], 32, 0, lane);
  #pragma unroll
  for (int nt = 0; nt < 8; ++nt) {
    v8f acc = {};
    acc = wmma_bf16(a0, bfrag(W1p, 8, 0, nt, lane), acc);
    int col = nt * 16 + (lane & 15);
    float bias = b1[col];
    int rbase = (lane >> 4) * 8;
    #pragma unroll
    for (int v = 0; v < 8; ++v)
      hidTile[wave][(rbase + v) * 128 + col] = f2bf(fmaxf(acc[v] + bias, 0.0f));
  }
  lds_fence();

  // layer 2: K=128 (4 ktiles), N=128 (8 ntiles)
  #pragma unroll
  for (int nt = 0; nt < 8; ++nt) {
    v8f acc = {};
    #pragma unroll
    for (int kt = 0; kt < 4; ++kt)
      acc = wmma_bf16(afrag_lds(hidTile[wave], 128, kt, lane),
                      bfrag(W2p, 8, kt, nt, lane), acc);
    int col = nt * 16 + (lane & 15);
    float bias = b2[col];
    int rbase = (lane >> 4) * 8;
    #pragma unroll
    for (int v = 0; v < 8; ++v) {
      int r = base + rbase + v;
      if (r < nNodes) hv[(size_t)r * HID + col] = f2bf(acc[v] + bias);
    }
  }
}

// ---------------------------------------------------------------------------
// edge_encoder + scatter: h_e = mlp2(concat(hv[row], hv[col], attr));
// agg[col] += h_e. One wave per 32-edge tile (two 16-row M-tiles share each
// B fragment -> half the B-operand traffic per FLOP). K padded 257 -> 288.
// ---------------------------------------------------------------------------
__global__ void __launch_bounds__(64) edge_kernel(
    const unsigned short* __restrict__ hv, const int* __restrict__ edge_index,
    const float* __restrict__ edge_attr, const float* __restrict__ b1,
    const float* __restrict__ b2, const unsigned short* __restrict__ W1p,
    const unsigned short* __restrict__ W2p, float* __restrict__ agg,
    int nEdges) {
  __shared__ unsigned short inTile[WPB_E][32 * 288];
  __shared__ unsigned short hidTile[WPB_E][32 * 128];
  __shared__ int colSh[WPB_E][32];
  int wave = threadIdx.x >> 5, lane = threadIdx.x & 31;
  int tile = blockIdx.x * WPB_E + wave;
  if (tile * 32 >= nEdges) return;
  int ebase = tile * 32;

  { // gather: each lane builds one full input row (both endpoints + attr)
    int e = ebase + lane;
    unsigned short* rowp = inTile[wave] + lane * 288;
    if (e < nEdges) {
      int srcIdx = edge_index[e];             // edge_index[0][e]
      int dstIdx = edge_index[nEdges + e];    // edge_index[1][e]
      colSh[wave][lane] = dstIdx;
      const v4u* s0 = (const v4u*)(hv + (size_t)srcIdx * HID);
      const v4u* s1 = (const v4u*)(hv + (size_t)dstIdx * HID);
      v4u* d0 = (v4u*)(rowp);
      v4u* d1 = (v4u*)(rowp + 128);
      #pragma unroll
      for (int j = 0; j < 16; ++j) d0[j] = s0[j];
      #pragma unroll
      for (int j = 0; j < 16; ++j) d1[j] = s1[j];
      rowp[256] = f2bf(edge_attr[e]);
      #pragma unroll
      for (int j = 257; j < 288; ++j) rowp[j] = 0;
    } else {
      colSh[wave][lane] = 0;
      v4u z = (v4u)(0u);
      v4u* d = (v4u*)(rowp);
      #pragma unroll
      for (int j = 0; j < 36; ++j) d[j] = z;   // 288 shorts = 36 v4u
    }
  }
  lds_fence();

  // layer 1: K=288 (9 ktiles), N=128 (8 ntiles), M=32 -> 144 WMMAs,
  // each B fragment loaded once and used by both M-tiles.
  #pragma unroll
  for (int nt = 0; nt < 8; ++nt) {
    v8f acc0 = {}, acc1 = {};
    #pragma unroll
    for (int kt = 0; kt < 9; ++kt) {
      v16bf b = bfrag(W1p, 8, kt, nt, lane);
      acc0 = wmma_bf16(afrag_lds(inTile[wave],            288, kt, lane), b, acc0);
      acc1 = wmma_bf16(afrag_lds(inTile[wave] + 16 * 288, 288, kt, lane), b, acc1);
    }
    int col = nt * 16 + (lane & 15);
    float bias = b1[col];
    int rbase = (lane >> 4) * 8;
    #pragma unroll
    for (int v = 0; v < 8; ++v) {
      hidTile[wave][(rbase + v) * 128 + col] =
          f2bf(fmaxf(acc0[v] + bias, 0.0f));
      hidTile[wave][(16 + rbase + v) * 128 + col] =
          f2bf(fmaxf(acc1[v] + bias, 0.0f));
    }
  }
  lds_fence();

  // layer 2: K=128 (4 ktiles), M=32 -> 64 WMMAs + fp32 atomic scatter
  #pragma unroll
  for (int nt = 0; nt < 8; ++nt) {
    v8f acc0 = {}, acc1 = {};
    #pragma unroll
    for (int kt = 0; kt < 4; ++kt) {
      v16bf b = bfrag(W2p, 8, kt, nt, lane);
      acc0 = wmma_bf16(afrag_lds(hidTile[wave],            128, kt, lane), b, acc0);
      acc1 = wmma_bf16(afrag_lds(hidTile[wave] + 16 * 128, 128, kt, lane), b, acc1);
    }
    int col = nt * 16 + (lane & 15);
    float bias = b2[col];
    int rbase = (lane >> 4) * 8;
    #pragma unroll
    for (int v = 0; v < 8; ++v) {
      int e0 = ebase + rbase + v;
      if (e0 < nEdges)
        atomicAdd(agg + (size_t)colSh[wave][rbase + v] * HID + col,
                  acc0[v] + bias);
      int e1 = ebase + 16 + rbase + v;
      if (e1 < nEdges)
        atomicAdd(agg + (size_t)colSh[wave][16 + rbase + v] * HID + col,
                  acc1[v] + bias);
    }
  }
}

// ---------------------------------------------------------------------------
// node_updater: out = relu(concat(hv, agg) @ W1 + b1) @ W2 + b2  -> [N,6] f32
// ---------------------------------------------------------------------------
__global__ void __launch_bounds__(128) node_updater_kernel(
    const unsigned short* __restrict__ hv, const float* __restrict__ agg,
    const float* __restrict__ b1, const float* __restrict__ b2,
    const unsigned short* __restrict__ W1p, const unsigned short* __restrict__ W2p,
    float* __restrict__ out, int nNodes) {
  __shared__ unsigned short inTile[WPB][16 * 256];
  __shared__ unsigned short hidTile[WPB][16 * 128];
  int wave = threadIdx.x >> 5, lane = threadIdx.x & 31;
  int tile = blockIdx.x * WPB + wave;
  if (tile * 16 >= nNodes) return;
  int base = tile * 16;

  { // build 16x256 bf16 input tile: cols 0..127 = hv row, 128..255 = bf16(agg row)
    int r = lane & 15, side = lane >> 4;
    int node = base + r;
    if (!side) {
      v4u* dst = (v4u*)(inTile[wave] + r * 256);
      if (node < nNodes) {
        const v4u* src = (const v4u*)(hv + (size_t)node * HID);
        #pragma unroll
        for (int j = 0; j < 16; ++j) dst[j] = src[j];
      } else {
        v4u z = (v4u)(0u);
        #pragma unroll
        for (int j = 0; j < 16; ++j) dst[j] = z;
      }
    } else {
      unsigned short* dst = inTile[wave] + r * 256 + 128;
      if (node < nNodes) {
        const float* src = agg + (size_t)node * HID;
        #pragma unroll 8
        for (int j = 0; j < 128; ++j) dst[j] = f2bf(src[j]);
      } else {
        #pragma unroll 8
        for (int j = 0; j < 128; ++j) dst[j] = 0;
      }
    }
  }
  lds_fence();

  // layer 1: K=256 (8 ktiles), N=128 (8 ntiles) -> 64 WMMAs
  #pragma unroll
  for (int nt = 0; nt < 8; ++nt) {
    v8f acc = {};
    #pragma unroll
    for (int kt = 0; kt < 8; ++kt)
      acc = wmma_bf16(afrag_lds(inTile[wave], 256, kt, lane),
                      bfrag(W1p, 8, kt, nt, lane), acc);
    int col = nt * 16 + (lane & 15);
    float bias = b1[col];
    int rbase = (lane >> 4) * 8;
    #pragma unroll
    for (int v = 0; v < 8; ++v)
      hidTile[wave][(rbase + v) * 128 + col] = f2bf(fmaxf(acc[v] + bias, 0.0f));
  }
  lds_fence();

  // layer 2: K=128 (4 ktiles), N=16 (1 ntile, 6 real cols) -> 4 WMMAs
  {
    v8f acc = {};
    #pragma unroll
    for (int kt = 0; kt < 4; ++kt)
      acc = wmma_bf16(afrag_lds(hidTile[wave], 128, kt, lane),
                      bfrag(W2p, 1, kt, 0, lane), acc);
    int col = lane & 15;
    float bias = (col < 6) ? b2[col] : 0.0f;
    int rbase = (lane >> 4) * 8;
    #pragma unroll
    for (int v = 0; v < 8; ++v) {
      int r = base + rbase + v;
      if (col < 6 && r < nNodes) out[(size_t)r * 6 + col] = acc[v] + bias;
    }
  }
}

// ---------------------------------------------------------------------------
static inline void launch_pack(const float* W, int K, int N, int rowStride,
                               int Kpad, int Npad, unsigned short* dst,
                               hipStream_t s) {
  int total = Kpad * Npad;
  int blocks = (total + 255) / 256;
  pack_weights_kernel<<<blocks, 256, 0, s>>>(W, K, N, rowStride, Kpad, Npad, dst);
}

extern "C" void kernel_launch(void* const* d_in, const int* in_sizes, int n_in,
                              void* d_out, int out_size, void* d_ws, size_t ws_size,
                              hipStream_t stream) {
  const float* node_states = (const float*)d_in[0];
  const int*   edge_index  = (const int*)d_in[1];
  const float* edge_attr   = (const float*)d_in[2];
  const float* ne_W1 = (const float*)d_in[3];
  const float* ne_b1 = (const float*)d_in[4];
  const float* ne_W2 = (const float*)d_in[5];
  const float* ne_b2 = (const float*)d_in[6];
  const float* ee_W1 = (const float*)d_in[7];
  const float* ee_b1 = (const float*)d_in[8];
  const float* ee_W2 = (const float*)d_in[9];
  const float* ee_b2 = (const float*)d_in[10];
  const float* nu_W1 = (const float*)d_in[11];
  const float* nu_b1 = (const float*)d_in[12];
  const float* nu_W2 = (const float*)d_in[13];
  const float* nu_b2 = (const float*)d_in[14];

  const int nNodes = in_sizes[0] / 9;
  const int nEdges = in_sizes[2];

  // Workspace layout (all 256B-aligned):
  char* ws = (char*)d_ws;
  size_t off = 0;
  auto alloc = [&](size_t bytes) {
    char* p = ws + off;
    off = (off + bytes + 255) & ~(size_t)255;
    return p;
  };
  unsigned short* hv  = (unsigned short*)alloc((size_t)nNodes * HID * 2);
  float*          agg = (float*)alloc((size_t)nNodes * HID * 4);
  unsigned short* neW1p = (unsigned short*)alloc(1 * 8 * 1024);   //  32x128
  unsigned short* neW2p = (unsigned short*)alloc(4 * 8 * 1024);   // 128x128
  unsigned short* eeW1p = (unsigned short*)alloc(9 * 8 * 1024);   // 288x128
  unsigned short* eeW2p = (unsigned short*)alloc(4 * 8 * 1024);   // 128x128
  unsigned short* nuW1p = (unsigned short*)alloc(8 * 8 * 1024);   // 256x128
  unsigned short* nuW2p = (unsigned short*)alloc(4 * 1 * 1024);   // 128x16
  (void)ws_size;

  // 1) pack weights into WMMA B-fragment layout (cheap, deterministic)
  launch_pack(ne_W1,   9, 128, 128,  32, 128, neW1p, stream);
  launch_pack(ne_W2, 128, 128, 128, 128, 128, neW2p, stream);
  launch_pack(ee_W1, 257, 128, 128, 288, 128, eeW1p, stream);
  launch_pack(ee_W2, 128, 128, 128, 128, 128, eeW2p, stream);
  launch_pack(nu_W1, 256, 128, 128, 256, 128, nuW1p, stream);
  launch_pack(nu_W2, 128,   6,   6, 128,  16, nuW2p, stream);

  // 2) zero the aggregation buffer (graph-capturable)
  hipMemsetAsync(agg, 0, (size_t)nNodes * HID * 4, stream);

  // 3) node encoder -> hv (bf16)
  int nodeTiles = (nNodes + 15) / 16;
  int nodeBlocks = (nodeTiles + WPB - 1) / WPB;
  node_encoder_kernel<<<nodeBlocks, WPB * 32, 0, stream>>>(
      node_states, ne_b1, ne_b2, neW1p, neW2p, hv, nNodes);

  // 4) edge encoder + scatter-add -> agg (32-edge tiles, 2 waves/block)
  int edgeTiles = (nEdges + 31) / 32;
  int edgeBlocks = (edgeTiles + WPB_E - 1) / WPB_E;
  edge_kernel<<<edgeBlocks, WPB_E * 32, 0, stream>>>(
      hv, edge_index, edge_attr, ee_b1, ee_b2, eeW1p, eeW2p, agg, nEdges);

  // 5) node updater -> out [N,6] fp32
  node_updater_kernel<<<nodeBlocks, WPB * 32, 0, stream>>>(
      hv, agg, nu_b1, nu_b2, nuW1p, nuW2p, (float*)d_out, nNodes);
}
